// tNN_71365176590715
// MI455X (gfx1250) — matrix operand order repaired and checked by reading, not compiled, not run on hardware
//
#include <hip/hip_runtime.h>

// CDNA5 / gfx1250: wave32, WMMA 16x16x32 bf16 with f32 accumulate.
// t-product == block-circulant GEMM: OUT[784][65536] = relu(Abc @ X + bias)
// bf16x3 split precision (hi*hi + hi*lo + lo*hi) ~ f32 accuracy at bf16 rates.
// A pre-packed in WMMA fragment order in d_ws (L2-resident).
// X staged per 32-K chunk into LDS *in B-fragment order* (12-dword slots:
// 16B-aligned, bank-conflict-free ds_load_b128), double-buffered + software
// pipelined global loads.

typedef __attribute__((ext_vector_type(16))) __bf16 v16bf;
typedef v16bf v16bf_a16 __attribute__((aligned(16)));
typedef __attribute__((ext_vector_type(8)))  float  v8f;

#define LMN      28
#define KDIM     784          // 28*28
#define KCHUNKS  25           // K padded to 800 = 25 * 32
#define NCOLS    65536
#define BN       128          // N tile per workgroup (8 subtiles of 16)
#define WAVES    7            // 7 waves * 16 rows = BM 112; 49 = 7 * 7
#define MT_TOTAL 49           // 784 / 16
#define SLOT     12           // dwords per (s,lane) B-fragment slot (8 data + 4 pad)
#define NLOAD    10           // ceil(2048 / 224) staging slots per thread

__device__ __forceinline__ unsigned int f32_to_bf16_rne(float f) {
  unsigned int u = __float_as_uint(f);
  u += 0x7FFFu + ((u >> 16) & 1u);       // round-to-nearest-even
  return u >> 16;
}
__device__ __forceinline__ float bf16_to_f32(unsigned int h) {
  return __uint_as_float(h << 16);
}

// ---------------------------------------------------------------------------
// Kernel 1: build block-circulant A in WMMA A-fragment order, split hi/lo.
// dword index i = (((mt*25 + c)*32 + lane)*8 + d)
// ---------------------------------------------------------------------------
__global__ void build_afrag(const float* __restrict__ W,
                            unsigned int* __restrict__ ahi,
                            unsigned int* __restrict__ alo) {
  const int total = MT_TOTAL * KCHUNKS * 32 * 8;
  int i = blockIdx.x * blockDim.x + threadIdx.x;
  if (i >= total) return;
  int d    = i & 7;
  int lane = (i >> 3) & 31;
  int c    = (i >> 8) % KCHUNKS;
  int mt   = i / (256 * KCHUNKS);
  int mloc = lane & 15;
  int half = lane >> 4;
  int klocal = (d >> 2) * 16 + half * 8 + (d & 3) * 2;  // ISA 16-bit A 16x32 layout
  int kglob  = c * 32 + klocal;
  int row  = mt * 16 + mloc;        // 0..783
  int kout = row / LMN;
  int m    = row % LMN;
  unsigned int hw[2], lw[2];
#pragma unroll
  for (int e = 0; e < 2; ++e) {
    float v = 0.0f;
    int kk = kglob + e;
    if (kk < KDIM) {
      int j = kk / LMN, n = kk % LMN;
      int widx = (kout - j + LMN) % LMN;          // circulant index
      v = W[widx * (LMN * LMN) + m * LMN + n];
    }
    unsigned int h = f32_to_bf16_rne(v);
    unsigned int l = f32_to_bf16_rne(v - bf16_to_f32(h));
    hw[e] = h; lw[e] = l;
  }
  ahi[i] = hw[0] | (hw[1] << 16);
  alo[i] = lw[0] | (lw[1] << 16);
}

// ---------------------------------------------------------------------------
// Kernel 2: split-precision WMMA GEMM + bias + ReLU.
// 7 waves, tile 112(M) x 128(N); K loop: 25 chunks of 32, double-buffered LDS.
// LDS B layout: sB[buf][hi/lo][(s*32+lane)*12 + v], v=0..7 packed bf16 pairs.
//   -> one 32B contiguous, 16B-aligned, bank-conflict-free read per fragment.
// ---------------------------------------------------------------------------
__global__ __launch_bounds__(WAVES * 32)
void tprod_gemm(const float* __restrict__ X,
                const float* __restrict__ Bias,
                const unsigned int* __restrict__ AHi,
                const unsigned int* __restrict__ ALo,
                float* __restrict__ Out) {
  __shared__ unsigned int sB[2][2][8 * 32 * SLOT];   // 48 KB total

  const int tid  = threadIdx.x;
  const int lane = tid & 31;
  const int wave = tid >> 5;
  const int mt   = blockIdx.x * WAVES + wave;   // 0..48
  const int n0   = blockIdx.y * BN;

  v8f acc[8] = {};

  float r0[NLOAD], r1[NLOAD];

  // prefetch global X slab for a chunk into registers (no LDS dependence)
  auto gload = [&](int c) {
#pragma unroll
    for (int t = 0; t < NLOAD; ++t) {
      int i = tid + 224 * t;
      bool valid = (i < 16 * BN);
      int col = i & (BN - 1);
      int kk2 = i >> 7;
      int k   = c * 32 + kk2 * 2;
      size_t g = (size_t)k * NCOLS + (size_t)(n0 + col);
      r0[t] = (valid && k     < KDIM) ? X[g]         : 0.0f;
      r1[t] = (valid && k + 1 < KDIM) ? X[g + NCOLS] : 0.0f;
    }
  };
  // split to bf16 hi/lo pairs and store in B-fragment order
  auto cstore = [&](int buf) {
#pragma unroll
    for (int t = 0; t < NLOAD; ++t) {
      int i = tid + 224 * t;
      if (i < 16 * BN) {
        int col = i & (BN - 1);
        int kk2 = i >> 7;
        int s = col >> 4, n = col & 15;
        int half = (kk2 >> 2) & 1, v = (kk2 >> 3) * 4 + (kk2 & 3);
        int addr = (s * 32 + half * 16 + n) * SLOT + v;
        unsigned int h0 = f32_to_bf16_rne(r0[t]);
        unsigned int h1 = f32_to_bf16_rne(r1[t]);
        unsigned int l0 = f32_to_bf16_rne(r0[t] - bf16_to_f32(h0));
        unsigned int l1 = f32_to_bf16_rne(r1[t] - bf16_to_f32(h1));
        sB[buf][0][addr] = h0 | (h1 << 16);
        sB[buf][1][addr] = l0 | (l1 << 16);
      }
    }
  };

  const unsigned int* aBase = AHi + (size_t)(mt * KCHUNKS) * 32 * 8;
  const unsigned int* aBaseLo = ALo + (size_t)(mt * KCHUNKS) * 32 * 8;

  gload(0);
  for (int c = 0; c < KCHUNKS; ++c) {
    const int buf = c & 1;
    cstore(buf);
    // A fragments: issue loads before the barrier so latency hides under it
    v16bf ahiF = *(const v16bf*)(aBase   + (c * 32 + lane) * 8);
    v16bf aloF = *(const v16bf*)(aBaseLo + (c * 32 + lane) * 8);
    __syncthreads();
    if (c + 1 < KCHUNKS) gload(c + 1);   // overlap next slab with WMMA burst

#pragma unroll
    for (int s = 0; s < 8; ++s) {
      v16bf bh = *(const v16bf_a16*)&sB[buf][0][(s * 32 + lane) * SLOT];
      v16bf bl = *(const v16bf_a16*)&sB[buf][1][(s * 32 + lane) * SLOT];
      acc[s] = __builtin_amdgcn_wmma_f32_16x16x32_bf16(
                   false, ahiF, false, bh, (short)0, acc[s], false, false);
      acc[s] = __builtin_amdgcn_wmma_f32_16x16x32_bf16(
                   false, ahiF, false, bl, (short)0, acc[s], false, false);
      acc[s] = __builtin_amdgcn_wmma_f32_16x16x32_bf16(
                   false, aloF, false, bh, (short)0, acc[s], false, false);
    }
  }

  // epilogue: bias + ReLU; C/D layout: VGPR r -> M = r (+8 for lanes>=16)
  const int colb  = lane & 15;
  const int mbase = mt * 16 + (lane >> 4) * 8;
#pragma unroll
  for (int s = 0; s < 8; ++s) {
    int col = n0 + s * 16 + colb;
#pragma unroll
    for (int r = 0; r < 8; ++r) {
      int row = mbase + r;
      float v = acc[s][r] + Bias[row];
      Out[(size_t)row * NCOLS + col] = fmaxf(v, 0.0f);
    }
  }
}

// ---------------------------------------------------------------------------
extern "C" void kernel_launch(void* const* d_in, const int* in_sizes, int n_in,
                              void* d_out, int out_size, void* d_ws, size_t ws_size,
                              hipStream_t stream) {
  const float* x = (const float*)d_in[0];   // (28,28,65536) == X[784][65536]
  const float* W = (const float*)d_in[1];   // (28,28,28)
  const float* B = (const float*)d_in[2];   // (28,28,1)  == bias[784]
  float* out = (float*)d_out;               // (28,28,65536) == OUT[784][65536]

  const int afrag_dwords = MT_TOTAL * KCHUNKS * 32 * 8;   // 313,600 dwords
  unsigned int* ahi = (unsigned int*)d_ws;
  unsigned int* alo = ahi + afrag_dwords;                  // ~2.5 MB total

  build_afrag<<<(afrag_dwords + 255) / 256, 256, 0, stream>>>(W, ahi, alo);

  dim3 grid(MT_TOTAL / WAVES, NCOLS / BN);   // (7, 512); x fastest -> X L2 reuse
  tprod_gemm<<<grid, WAVES * 32, 0, stream>>>(x, B, ahi, alo, out);
}
